// GreedyBatchedRNNTLoopLabelsComputer_46883863003213
// MI455X (gfx1250) — matrix operand order repaired and checked by reading, not compiled, not run on hardware
//
#include <hip/hip_runtime.h>
#include <hip/hip_bf16.h>

// ---------------- problem constants ----------------
#define B_      64
#define T_      128
#define ENC_    512
#define P_      320
#define H_      320
#define J_      320
#define V_      1025
#define VP_     1040          // V padded to multiple of 16
#define G4H_    1280          // 4*H
#define KG_     640           // K for gates GEMM ([emb|h])
#define BLANK_  1024
#define MAXSYM_ 2
#define L_      256           // T*MAX_SYMBOLS
#define STEPS_  384           // T*(MAX_SYMBOLS+1)

// ---------------- WMMA types / helpers ----------------
typedef __attribute__((ext_vector_type(16))) __bf16 v16bf;
typedef __attribute__((ext_vector_type(8)))  __bf16 v8bf;
typedef __attribute__((ext_vector_type(8)))  float  v8f;

static __device__ __forceinline__ __bf16 f2bf(float f) {
  unsigned u = __builtin_bit_cast(unsigned, f);
  u += 0x7FFFu + ((u >> 16) & 1u);            // round-to-nearest-even
  unsigned short hsh = (unsigned short)(u >> 16);
  return __builtin_bit_cast(__bf16, hsh);
}

static __device__ __forceinline__ float sigf(float x) {
  return 1.0f / (1.0f + __expf(-x));
}

// A operand (16x32 bf16): lane needs k = {8*khalf+0..7} and {16+8*khalf+0..7}.
// p must already include the 8*khalf element offset. Two 16B loads.
static __device__ __forceinline__ v16bf load_a32(const __bf16* p) {
  v8bf lo = *(const v8bf*)p;
  v8bf hi = *(const v8bf*)(p + 16);
  v16bf r;
#pragma unroll
  for (int i = 0; i < 8; ++i) { r[i] = lo[i]; r[i + 8] = hi[i]; }
  return r;
}

// B operand (32x16 bf16): lane needs k = {8*khalf+0..7} and {16+8*khalf+0..7}.
// p must already include the 8*khalf element offset. Two 16B loads.
static __device__ __forceinline__ v16bf load_b32(const __bf16* p) {
  v8bf lo = *(const v8bf*)p;
  v8bf hi = *(const v8bf*)(p + 16);
  v16bf r;
#pragma unroll
  for (int i = 0; i < 8; ++i) { r[i] = lo[i]; r[i + 8] = hi[i]; }
  return r;
}

#define WMMA_BF16(a, b, c) \
  __builtin_amdgcn_wmma_f32_16x16x32_bf16(false, (a), false, (b), (short)0, (c), false, false)

// ---------------- setup kernels ----------------
__global__ void k_cvt(const float* __restrict__ src, __bf16* __restrict__ dst, int n) {
  int i = blockIdx.x * blockDim.x + threadIdx.x;
  if (i < n) dst[i] = f2bf(src[i]);
}

__global__ void k_cvt_pad(const float* __restrict__ src, __bf16* __restrict__ dst,
                          int rows, int rowspad, int cols) {
  int i = blockIdx.x * blockDim.x + threadIdx.x;
  if (i >= rowspad * cols) return;
  int r = i / cols, cc = i - r * cols;
  dst[i] = f2bf(r < rows ? src[r * cols + cc] : 0.0f);
}

__global__ void k_init(float* h, float* c, float* pp, __bf16* hbf,
                       int* st_t, int* st_sym, int* st_ll, int* st_nd,
                       int* hyp_l, int* hyp_t, int* lens, float* scores) {
  int i = blockIdx.x * blockDim.x + threadIdx.x;
  if (i < B_ * H_) { h[i] = 0.0f; c[i] = 0.0f; hbf[i] = f2bf(0.0f); }
  if (i < B_ * J_) pp[i] = 0.0f;
  if (i < B_ * L_) { hyp_l[i] = 0; hyp_t[i] = 0; }
  if (i < B_) {
    st_t[i] = 0; st_sym[i] = 0; st_ll[i] = BLANK_; st_nd[i] = 1;
    lens[i] = 0; scores[i] = 0.0f;
  }
}

// ---------------- encoder projection: x_proj = x @ W_enc^T + b_enc ----------------
// M = 8192, N = 320, K = 512.  grid = (5, 128), block = 128 (4 waves).
// Each block stages a 64-row slab of x in LDS (64 KB); waves cover 4 n-tiles.
__global__ void k_encproj(const __bf16* __restrict__ xbf, const __bf16* __restrict__ Wencbf,
                          const float* __restrict__ benc, float* __restrict__ xproj) {
  extern __shared__ __bf16 AsE[];               // [64][512]
  const int wave = threadIdx.x >> 5;
  const int lane = threadIdx.x & 31;
  const int g  = blockIdx.y;                    // row group, 0..127
  const int nt = blockIdx.x * 4 + wave;         // 0..19

  // stage A slab: 64 x 512 bf16 = 64 KB, uint4 chunks of 8 bf16
  const int CPR = ENC_ / 8;                     // 64 chunks per row
  for (int cidx = threadIdx.x; cidx < 64 * CPR; cidx += blockDim.x) {
    int row = cidx / CPR, cc = cidx - row * CPR;
    *(uint4*)(AsE + row * ENC_ + cc * 8) =
        *(const uint4*)(xbf + ((size_t)(g * 64 + row)) * ENC_ + cc * 8);
  }
  __syncthreads();

  const int khalf = lane >> 4;
  const int nB = nt * 16 + (lane & 15);
  v8f acc[4] = {};
  for (int kk = 0; kk < ENC_; kk += 32) {
    const __bf16* bp = Wencbf + (size_t)nB * ENC_ + kk + 8 * khalf;
    __builtin_prefetch(bp + 64, 0, 1);          // global_prefetch_b8
    v16bf b = load_b32(bp);
#pragma unroll
    for (int m2 = 0; m2 < 4; ++m2) {
      v16bf a = load_a32(AsE + (m2 * 16 + (lane & 15)) * ENC_ + kk + 8 * khalf);
      acc[m2] = WMMA_BF16(a, b, acc[m2]);
    }
  }
  const float bias = benc[nB];
#pragma unroll
  for (int m2 = 0; m2 < 4; ++m2) {
#pragma unroll
    for (int r = 0; r < 8; ++r) {
      const int m = g * 64 + m2 * 16 + r + 8 * khalf;
      xproj[(size_t)m * J_ + nB] = acc[m2][r] + bias;
    }
  }
}

// ---------------- step kernel 1: gates = [emb(ll) | h] @ [W_ih | W_hh]^T ----------------
// M = 64, N = 1280 (80 n-tiles), K = 640.  grid = 10, block = 256 (8 waves).
// Block stages the gathered [emb|h] matrix (64 x 640 bf16 = 80 KB) in LDS once.
__global__ void k_gates(const __bf16* __restrict__ embbf, const __bf16* __restrict__ hbf,
                        const __bf16* __restrict__ Wihbf, const __bf16* __restrict__ Whhbf,
                        const int* __restrict__ last_label, float* __restrict__ gates) {
  extern __shared__ __bf16 AsG[];               // [64][640]
  const int wave = threadIdx.x >> 5;
  const int lane = threadIdx.x & 31;
  const int nt = blockIdx.x * 8 + wave;         // 0..79

  const int CPR = KG_ / 8;                      // 80 chunks per row
  for (int cidx = threadIdx.x; cidx < 64 * CPR; cidx += blockDim.x) {
    int row = cidx / CPR, cc = cidx - row * CPR;
    uint4 v;
    if (cc < P_ / 8) {
      int llr = last_label[row];
      v = *(const uint4*)(embbf + (size_t)llr * P_ + cc * 8);
    } else {
      v = *(const uint4*)(hbf + (size_t)row * H_ + (cc - P_ / 8) * 8);
    }
    *(uint4*)(AsG + row * KG_ + cc * 8) = v;
  }
  __syncthreads();

  const int khalf = lane >> 4;
  const int n = nt * 16 + (lane & 15);
  v8f acc[4] = {};
  for (int kk = 0; kk < KG_; kk += 32) {
    const __bf16* bp = (kk < H_) ? (Wihbf + (size_t)n * P_ + kk)
                                 : (Whhbf + (size_t)n * H_ + (kk - H_));
    bp += 8 * khalf;
    __builtin_prefetch(bp + 64, 0, 1);
    v16bf b = load_b32(bp);
#pragma unroll
    for (int m2 = 0; m2 < 4; ++m2) {
      v16bf a = load_a32(AsG + (m2 * 16 + (lane & 15)) * KG_ + kk + 8 * khalf);
      acc[m2] = WMMA_BF16(a, b, acc[m2]);
    }
  }
#pragma unroll
  for (int m2 = 0; m2 < 4; ++m2) {
#pragma unroll
    for (int r = 0; r < 8; ++r) {
      const int m = m2 * 16 + r + 8 * khalf;
      gates[(size_t)m * G4H_ + n] = acc[m2][r];
    }
  }
}

// ---------------- step kernel 2: LSTM cell pointwise + masked state update ----------------
__global__ void k_cell(const float* __restrict__ gates, const float* __restrict__ bih,
                       const float* __restrict__ bhh, const int* __restrict__ out_len,
                       const int* __restrict__ st_t, const int* __restrict__ st_nd,
                       float* __restrict__ h, float* __restrict__ c,
                       __bf16* __restrict__ hbf, __bf16* __restrict__ hnewbf) {
  int idx = blockIdx.x * blockDim.x + threadIdx.x;
  if (idx >= B_ * H_) return;
  int b = idx / H_, j = idx - b * H_;
  float ig = gates[(size_t)b * G4H_ + j]          + bih[j]          + bhh[j];
  float fg = gates[(size_t)b * G4H_ + H_ + j]     + bih[H_ + j]     + bhh[H_ + j];
  float gg = gates[(size_t)b * G4H_ + 2 * H_ + j] + bih[2 * H_ + j] + bhh[2 * H_ + j];
  float og = gates[(size_t)b * G4H_ + 3 * H_ + j] + bih[3 * H_ + j] + bhh[3 * H_ + j];
  float cn = sigf(fg) * c[idx] + sigf(ig) * tanhf(gg);
  float hn = sigf(og) * tanhf(cn);
  bool upd = (st_nd[b] != 0) && (st_t[b] < out_len[b]);
  float hs = upd ? hn : h[idx];
  float cs = upd ? cn : c[idx];
  h[idx] = hs;
  c[idx] = cs;
  hbf[idx]    = f2bf(hs);   // masked h for next step's gates GEMM
  hnewbf[idx] = f2bf(hn);   // unmasked h_new for pred-proj GEMM
}

// ---------------- step kernel 3: pred_proj update + joint relu operand ----------------
// M = 64, N = 320 (20 n-tiles), K = 320.  grid = 2, block = 320 (10 waves).
__global__ void k_pred(const __bf16* __restrict__ hnewbf, const __bf16* __restrict__ Wpredbf,
                       const float* __restrict__ bpred, const float* __restrict__ xproj,
                       const int* __restrict__ out_len, const int* __restrict__ st_t,
                       const int* __restrict__ st_nd, float* __restrict__ pp,
                       __bf16* __restrict__ rjbf) {
  extern __shared__ __bf16 AsP[];               // [64][320]
  const int wave = threadIdx.x >> 5;
  const int lane = threadIdx.x & 31;
  const int nt = blockIdx.x * 10 + wave;        // 0..19

  const int CPR = H_ / 8;                       // 40 chunks per row
  for (int cidx = threadIdx.x; cidx < 64 * CPR; cidx += blockDim.x) {
    int row = cidx / CPR, cc = cidx - row * CPR;
    *(uint4*)(AsP + row * H_ + cc * 8) = *(const uint4*)(hnewbf + (size_t)row * H_ + cc * 8);
  }
  __syncthreads();

  const int khalf = lane >> 4;
  const int n = nt * 16 + (lane & 15);
  v8f acc[4] = {};
  for (int kk = 0; kk < H_; kk += 32) {
    const __bf16* bp = Wpredbf + (size_t)n * H_ + kk + 8 * khalf;
    __builtin_prefetch(bp + 64, 0, 1);
    v16bf b = load_b32(bp);
#pragma unroll
    for (int m2 = 0; m2 < 4; ++m2) {
      v16bf a = load_a32(AsP + (m2 * 16 + (lane & 15)) * H_ + kk + 8 * khalf);
      acc[m2] = WMMA_BF16(a, b, acc[m2]);
    }
  }
  const float bias = bpred[n];
#pragma unroll
  for (int m2 = 0; m2 < 4; ++m2) {
#pragma unroll
    for (int r = 0; r < 8; ++r) {
      const int b = m2 * 16 + r + 8 * khalf;
      float ppnew = acc[m2][r] + bias;
      int  t  = st_t[b];
      int  ol = out_len[b];
      bool upd = (st_nd[b] != 0) && (t < ol);
      float ppv = upd ? ppnew : pp[(size_t)b * J_ + n];
      pp[(size_t)b * J_ + n] = ppv;
      int safe = (t < ol - 1) ? t : (ol - 1);   // min(t, out_len-1)
      float rv = xproj[((size_t)b * T_ + safe) * J_ + n] + ppv;
      rjbf[(size_t)b * J_ + n] = f2bf(rv > 0.0f ? rv : 0.0f);
    }
  }
}

// ---------------- step kernel 4: logits = relu(enc+pp) @ W_out^T + b_out ----------------
// M = 64, N = 1040 (65 n-tiles), K = 320.  grid = 5, block = 416 (13 waves): exact.
__global__ void k_logits(const __bf16* __restrict__ rjbf, const __bf16* __restrict__ Woutbf,
                         const float* __restrict__ bout, float* __restrict__ logits) {
  extern __shared__ __bf16 AsL[];               // [64][320]
  const int wave = threadIdx.x >> 5;
  const int lane = threadIdx.x & 31;
  const int nt = blockIdx.x * 13 + wave;        // 0..64

  const int CPR = J_ / 8;                       // 40 chunks per row
  for (int cidx = threadIdx.x; cidx < 64 * CPR; cidx += blockDim.x) {
    int row = cidx / CPR, cc = cidx - row * CPR;
    *(uint4*)(AsL + row * J_ + cc * 8) = *(const uint4*)(rjbf + (size_t)row * J_ + cc * 8);
  }
  __syncthreads();

  const int khalf = lane >> 4;
  const int n = nt * 16 + (lane & 15);
  v8f acc[4] = {};
  for (int kk = 0; kk < J_; kk += 32) {
    const __bf16* bp = Woutbf + (size_t)n * J_ + kk + 8 * khalf;
    __builtin_prefetch(bp + 64, 0, 1);
    v16bf b = load_b32(bp);
#pragma unroll
    for (int m2 = 0; m2 < 4; ++m2) {
      v16bf a = load_a32(AsL + (m2 * 16 + (lane & 15)) * J_ + kk + 8 * khalf);
      acc[m2] = WMMA_BF16(a, b, acc[m2]);
    }
  }
  const float bias = (n < V_) ? bout[n] : 0.0f;
#pragma unroll
  for (int m2 = 0; m2 < 4; ++m2) {
#pragma unroll
    for (int r = 0; r < 8; ++r) {
      const int b = m2 * 16 + r + 8 * khalf;
      float val = (n < V_) ? (acc[m2][r] + bias) : -3.0e38f;
      logits[(size_t)b * VP_ + n] = val;
    }
  }
}

// ---------------- step kernel 5: per-row argmax + hypothesis bookkeeping ----------------
__global__ void k_finalize(const float* __restrict__ logits, const int* __restrict__ out_len,
                           int* st_t, int* st_sym, int* st_ll, int* st_nd,
                           int* hyp_l, int* hyp_t, int* lens, float* scores) {
  const int wv   = (blockIdx.x * blockDim.x + threadIdx.x) >> 5;
  const int lane = threadIdx.x & 31;
  if (wv >= B_) return;
  const int b = wv;

  float mv = -3.4e38f;
  int   mi = 0;
  for (int n = lane; n < VP_; n += 32) {
    float v = logits[(size_t)b * VP_ + n];
    if (v > mv) { mv = v; mi = n; }             // keeps first max within stride
  }
#pragma unroll
  for (int off = 16; off > 0; off >>= 1) {
    float ov = __shfl_xor(mv, off, 32);
    int   oi = __shfl_xor(mi, off, 32);
    if (ov > mv || (ov == mv && oi < mi)) { mv = ov; mi = oi; }
  }

  if (lane == 0) {
    int  t   = st_t[b];
    int  sym = st_sym[b];
    bool active   = t < out_len[b];
    bool is_blank = (mi == BLANK_);
    bool emit     = active && !is_blank;
    int  len = lens[b];
    if (emit) {
      int widx = (len < L_ - 1) ? len : (L_ - 1);
      hyp_l[b * L_ + widx] = mi;
      hyp_t[b * L_ + widx] = t;
      scores[b] += mv;
      len += 1;
    }
    int  sym_new = emit ? (sym + 1) : sym;
    bool advance = active && (is_blank || (sym_new >= MAXSYM_));
    st_t[b]   = t + (advance ? 1 : 0);
    st_sym[b] = advance ? 0 : sym_new;
    if (emit) st_ll[b] = mi;
    st_nd[b]  = emit ? 1 : 0;
    lens[b]   = len;
  }
}

// ---------------- host launcher ----------------
extern "C" void kernel_launch(void* const* d_in, const int* in_sizes, int n_in,
                              void* d_out, int out_size, void* d_ws, size_t ws_size,
                              hipStream_t stream) {
  (void)in_sizes; (void)n_in; (void)out_size; (void)ws_size;
  const float* x      = (const float*)d_in[0];
  const int*   olen   = (const int*)  d_in[1];
  const float* emb    = (const float*)d_in[2];
  const float* W_ih   = (const float*)d_in[3];
  const float* W_hh   = (const float*)d_in[4];
  const float* b_ih   = (const float*)d_in[5];
  const float* b_hh   = (const float*)d_in[6];
  const float* W_enc  = (const float*)d_in[7];
  const float* b_enc  = (const float*)d_in[8];
  const float* W_pred = (const float*)d_in[9];
  const float* b_pred = (const float*)d_in[10];
  const float* W_out  = (const float*)d_in[11];
  const float* b_out  = (const float*)d_in[12];

  char* ws = (char*)d_ws;
  size_t off = 0;
  auto alloc = [&](size_t bytes) -> void* {
    void* p = ws + off;
    off = (off + bytes + 255) & ~(size_t)255;
    return p;
  };

  __bf16* Wihbf   = (__bf16*)alloc((size_t)G4H_ * P_ * 2);
  __bf16* Whhbf   = (__bf16*)alloc((size_t)G4H_ * H_ * 2);
  __bf16* Wpredbf = (__bf16*)alloc((size_t)J_ * H_ * 2);
  __bf16* Woutbf  = (__bf16*)alloc((size_t)VP_ * J_ * 2);
  __bf16* embbf   = (__bf16*)alloc((size_t)V_ * P_ * 2);
  __bf16* Wencbf  = (__bf16*)alloc((size_t)J_ * ENC_ * 2);
  __bf16* xbf     = (__bf16*)alloc((size_t)B_ * T_ * ENC_ * 2);
  float*  xproj   = (float*) alloc((size_t)B_ * T_ * J_ * 4);
  float*  gates   = (float*) alloc((size_t)B_ * G4H_ * 4);
  float*  h       = (float*) alloc((size_t)B_ * H_ * 4);
  float*  c       = (float*) alloc((size_t)B_ * H_ * 4);
  float*  pp      = (float*) alloc((size_t)B_ * J_ * 4);
  __bf16* hbf     = (__bf16*)alloc((size_t)B_ * H_ * 2);
  __bf16* hnewbf  = (__bf16*)alloc((size_t)B_ * H_ * 2);
  __bf16* rjbf    = (__bf16*)alloc((size_t)B_ * J_ * 2);
  float*  logits  = (float*) alloc((size_t)B_ * VP_ * 4);
  int*    st_t    = (int*)   alloc(B_ * 4);
  int*    st_sym  = (int*)   alloc(B_ * 4);
  int*    st_ll   = (int*)   alloc(B_ * 4);
  int*    st_nd   = (int*)   alloc(B_ * 4);

  // output layout: hyp_l[64*256] i32, hyp_t[64*256] i32, lens[64] i32, scores[64] f32
  int*   hyp_l  = (int*)d_out;
  int*   hyp_t  = hyp_l + B_ * L_;
  int*   lens   = hyp_t + B_ * L_;
  float* scores = (float*)(lens + B_);

  // one-time (per call) conversions + init
  k_cvt<<<(G4H_ * P_ + 255) / 256, 256, 0, stream>>>(W_ih,  Wihbf,   G4H_ * P_);
  k_cvt<<<(G4H_ * H_ + 255) / 256, 256, 0, stream>>>(W_hh,  Whhbf,   G4H_ * H_);
  k_cvt<<<(J_ * H_ + 255) / 256,   256, 0, stream>>>(W_pred, Wpredbf, J_ * H_);
  k_cvt<<<(V_ * P_ + 255) / 256,   256, 0, stream>>>(emb,   embbf,   V_ * P_);
  k_cvt<<<(J_ * ENC_ + 255) / 256, 256, 0, stream>>>(W_enc, Wencbf,  J_ * ENC_);
  k_cvt<<<(B_ * T_ * ENC_ + 255) / 256, 256, 0, stream>>>(x, xbf, B_ * T_ * ENC_);
  k_cvt_pad<<<(VP_ * J_ + 255) / 256, 256, 0, stream>>>(W_out, Woutbf, V_, VP_, J_);
  k_init<<<(B_ * H_ + 255) / 256, 256, 0, stream>>>(h, c, pp, hbf, st_t, st_sym, st_ll,
                                                    st_nd, hyp_l, hyp_t, lens, scores);

  // encoder projection: (5 x 128) blocks of 4 waves; 64 KB LDS A-slab per block
  k_encproj<<<dim3(5, 128), 128, 64 * 1024, stream>>>(xbf, Wencbf, b_enc, xproj);

  // sequential greedy decode: 384 graph-captured steps
  for (int s = 0; s < STEPS_; ++s) {
    k_gates<<<10, 256, 80 * 1024, stream>>>(embbf, hbf, Wihbf, Whhbf, st_ll, gates);
    k_cell<<<(B_ * H_ + 255) / 256, 256, 0, stream>>>(gates, b_ih, b_hh, olen,
                                                      st_t, st_nd, h, c, hbf, hnewbf);
    k_pred<<<2, 320, 40 * 1024, stream>>>(hnewbf, Wpredbf, b_pred, xproj, olen,
                                          st_t, st_nd, pp, rjbf);
    k_logits<<<5, 416, 40 * 1024, stream>>>(rjbf, Woutbf, b_out, logits);
    k_finalize<<<8, 256, 0, stream>>>(logits, olen, st_t, st_sym, st_ll, st_nd,
                                      hyp_l, hyp_t, lens, scores);
  }
}
